// Attention_6631429505804
// MI455X (gfx1250) — matrix operand order repaired and checked by reading, not compiled, not run on hardware
//
#include <hip/hip_runtime.h>

typedef __attribute__((ext_vector_type(16))) __bf16 v16bf;
typedef __attribute__((ext_vector_type(8)))  float  v8f;

#define B_   2
#define S_   2048
#define H_   32
#define HK_  8
#define G_   (H_ / HK_)
#define D_   128
#define SCALE_ 0.08838834764831845f
#define LOG2E_ 1.4426950408889634f
#define NEGINF_ -1e30f

#define KSTR 136   // LDS row stride (halfs) for K tile [32 keys][128 d]  (272B, 16B-aligned)
#define VSTR 40    // LDS row stride (halfs) for V^T tile [128 d][32 keys] (80B, 16B-aligned)

union U16 {
  v16bf v;
  unsigned short s[16];
  uint4 q4[2];
};

// native bf16 convert (RNE fptrunc -> v_cvt_pk_bf16_f32 on gfx1250, verified)
__device__ __forceinline__ unsigned short f2bf(float f) {
  return __builtin_bit_cast(unsigned short, (__bf16)f);
}
__device__ __forceinline__ unsigned pk2(float x, float y) {
  return (unsigned)f2bf(x) | ((unsigned)f2bf(y) << 16);
}
__device__ __forceinline__ float fc(const float4& f, int c) {
  return c == 0 ? f.x : c == 1 ? f.y : c == 2 ? f.z : f.w;
}
// v_exp_f32 computes 2^x natively
__device__ __forceinline__ float exp2fast(float x) {
  return __builtin_amdgcn_exp2f(x);
}

__global__ __launch_bounds__(256)
void fa_fwd_kernel(const float* __restrict__ Qg,
                   const float* __restrict__ Kg,
                   const float* __restrict__ Vg,
                   float* __restrict__ Og) {
  __shared__ unsigned short ldsK[2][32 * KSTR];    // double-buffered K tile (bf16, [key][d])
  __shared__ unsigned short ldsV[2][128 * VSTR];   // double-buffered V^T tile (bf16, [d][key])

  const int tid  = threadIdx.x;
  const int lane = tid & 31;
  const int wave = tid >> 5;                    // 0..7
  const int l15  = lane & 15;
  const int hihalf = lane >> 4;                 // 0: lanes 0-15, 1: lanes 16-31

  const int qBlockBase = blockIdx.x * 128;      // 128 query rows per 8-wave block
  const int h  = blockIdx.y;
  const int b  = blockIdx.z;
  const int kh = h / G_;

  const int qWaveBase = qBlockBase + wave * 16;
  const int qRow = qWaveBase + l15;

  // ---- Q fragments: B-operand layout for S^T = K x Q^T.
  // SCALE * log2(e) folded in => scores live in the base-2 domain.
  U16 qf[4];
  {
    const float* qp = Qg + (((size_t)b * S_ + qRow) * H_ + h) * D_;
#pragma unroll
    for (int c = 0; c < 4; ++c) {
      const float4* p = (const float4*)(qp + c * 32 + hihalf * 8);
#pragma unroll
      for (int i = 0; i < 4; ++i) {
        float4 f = p[i + (i & 2)];
        qf[c].s[i * 4 + 0] = f2bf(f.x * (SCALE_ * LOG2E_));
        qf[c].s[i * 4 + 1] = f2bf(f.y * (SCALE_ * LOG2E_));
        qf[c].s[i * 4 + 2] = f2bf(f.z * (SCALE_ * LOG2E_));
        qf[c].s[i * 4 + 3] = f2bf(f.w * (SCALE_ * LOG2E_));
      }
    }
  }

  const v8f vzero = {};
  v8f acc[8];
#pragma unroll
  for (int dg = 0; dg < 8; ++dg) acc[dg] = vzero;
  float mRun = NEGINF_, lRun = 0.f;

  const int nTiles = qBlockBase / 32 + 4;       // causal: keys up to qBlockBase+127
  const size_t keyStride = (size_t)HK_ * D_;    // floats between consecutive keys

  // staging thread mappings (256 threads; each converts 16 K + 16 V floats/tile)
  const int kRow = tid >> 3;                    // K: key row 0..31
  const int kD0  = (tid & 7) * 16;              // K: 16-d chunk (8 threads sweep a row)
  const float* kBaseP = Kg + (((size_t)b * S_ + kRow) * HK_ + kh) * D_ + kD0;
  const int vK0 = (tid >> 5) * 4;               // V: 4-key group 0,4,...,28
  const int vD0 = (tid & 31) * 4;               // V: 4-d chunk (32 threads sweep a row)
  const float* vBaseP = Vg + (((size_t)b * S_ + vK0) * HK_ + kh) * D_ + vD0;

  for (int t = 0; t < nTiles; ++t) {
    const int keyBase = t * 32;
    const int buf = t & 1;

    // ---- stage tile t (bf16 convert on the fly), prefetch tile t+1
    {
      const float4* kp = (const float4*)(kBaseP + (size_t)keyBase * keyStride);
      unsigned short* dK = &ldsK[buf][kRow * KSTR + kD0];
#pragma unroll
      for (int i = 0; i < 2; ++i) {
        float4 f0 = kp[2 * i], f1 = kp[2 * i + 1];
        uint4 w;
        w.x = pk2(f0.x, f0.y); w.y = pk2(f0.z, f0.w);
        w.z = pk2(f1.x, f1.y); w.w = pk2(f1.z, f1.w);
        *(uint4*)(dK + i * 8) = w;               // ds_store_b128
      }
      const float* vp = vBaseP + (size_t)keyBase * keyStride;
      float4 va[4];
#pragma unroll
      for (int i = 0; i < 4; ++i)                // 4 consecutive keys, 4 d each
        va[i] = *(const float4*)(vp + i * keyStride);
#pragma unroll
      for (int j = 0; j < 4; ++j) {              // transpose-pack: 4 keys per b64 store
        uint2 w;
        w.x = pk2(fc(va[0], j), fc(va[1], j));
        w.y = pk2(fc(va[2], j), fc(va[3], j));
        *(uint2*)&ldsV[buf][(vD0 + j) * VSTR + vK0] = w;   // ds_store_b64
      }
      if (t + 1 < nTiles) {                      // gfx1250 global_prefetch_b8
        __builtin_prefetch(kBaseP + (size_t)(keyBase + 32) * keyStride, 0, 1);
        __builtin_prefetch(vBaseP + (size_t)(keyBase + 32) * keyStride, 0, 1);
      }
    }
    __syncthreads();   // single barrier per tile: double buffering covers WAR hazard

    if (keyBase > qWaveBase + 15) continue;      // wave-uniform: tile fully masked

    // ---- scores S^T = K_tile x Q^T : two 16(key)x16(query) C tiles
    v8f st0 = vzero, st1 = vzero;
#pragma unroll
    for (int c = 0; c < 4; ++c) {
      U16 a0, a1;
      {
        int o = l15 * KSTR + c * 32 + hihalf * 8;
        a0.q4[0] = *(const uint4*)&ldsK[buf][o];
        a0.q4[1] = *(const uint4*)&ldsK[buf][o + 16];
      }
      {
        int o = (16 + l15) * KSTR + c * 32 + hihalf * 8;
        a1.q4[0] = *(const uint4*)&ldsK[buf][o];
        a1.q4[1] = *(const uint4*)&ldsK[buf][o + 16];
      }
      st0 = __builtin_amdgcn_wmma_f32_16x16x32_bf16(false, a0.v, false, qf[c].v,
                                                    (short)0, st0, false, false);
      st1 = __builtin_amdgcn_wmma_f32_16x16x32_bf16(false, a1.v, false, qf[c].v,
                                                    (short)0, st1, false, false);
    }

    // ---- causal mask: only the boundary tiles per wave need it
    const int qIdx = qWaveBase + l15;
    const int kOfs = hihalf * 8;
    if (keyBase + 31 > qWaveBase) {              // wave-uniform branch
      const int rel0 = qIdx - keyBase - kOfs;    // st0[vv] valid iff vv <= rel0
      const int rel1 = rel0 - 16;
#pragma unroll
      for (int vv = 0; vv < 8; ++vv) {
        st0[vv] = (vv <= rel0) ? st0[vv] : NEGINF_;
        st1[vv] = (vv <= rel1) ? st1[vv] : NEGINF_;
      }
    }

    // ---- online softmax in base-2 domain (lane = query; vgpr/half = key)
    float mTile = NEGINF_;
#pragma unroll
    for (int vv = 0; vv < 8; ++vv)
      mTile = fmaxf(mTile, fmaxf(st0[vv], st1[vv]));
    mTile = fmaxf(mTile, __shfl_xor(mTile, 16, 32));
    float mNew  = fmaxf(mRun, mTile);
    float alpha = exp2fast(mRun - mNew);         // v_exp_f32, no pre-multiply
    mRun = mNew;

    float lTile = 0.f;
    U16 pa;                                      // P already in A-operand layout
#pragma unroll
    for (int vv = 0; vv < 8; ++vv) {
      float p0 = exp2fast(st0[vv] - mNew);
      float p1 = exp2fast(st1[vv] - mNew);
      lTile += p0 + p1;
      pa.s[vv]     = f2bf(p0);
      pa.s[vv + 8] = f2bf(p1);
    }
    lTile += __shfl_xor(lTile, 16, 32);
    lRun = lRun * alpha + lTile;

    // broadcast per-query alpha: acc element e is query q = e + 8*hihalf
    float asel[8];
#pragma unroll
    for (int e = 0; e < 8; ++e) {
      float aLo = __shfl(alpha, e, 32);
      float aHi = __shfl(alpha, e + 8, 32);
      asel[e] = hihalf ? aHi : aLo;
    }
#pragma unroll
    for (int dg = 0; dg < 8; ++dg) {
#pragma unroll
      for (int e = 0; e < 8; ++e) acc[dg][e] *= asel[e];
    }

    // ---- acc += P x V  (B-operand = V^T tile from LDS: lane = d column)
#pragma unroll
    for (int dg = 0; dg < 8; ++dg) {
      U16 bv;
      int o = (dg * 16 + l15) * VSTR + hihalf * 8;
      bv.q4[0] = *(const uint4*)&ldsV[buf][o];
      bv.q4[1] = *(const uint4*)&ldsV[buf][o + 16];
      acc[dg] = __builtin_amdgcn_wmma_f32_16x16x32_bf16(false, pa.v, false, bv.v,
                                                        (short)0, acc[dg], false, false);
    }
  }

  // ---- epilogue: 1/l per query, coalesced fp32 stores (lanes = consecutive d)
  float linv = 1.0f / lRun;
  float lsel[8];
#pragma unroll
  for (int e = 0; e < 8; ++e) {
    float lLo = __shfl(linv, e, 32);
    float lHi = __shfl(linv, e + 8, 32);
    lsel[e] = hihalf ? lHi : lLo;
  }
#pragma unroll
  for (int dg = 0; dg < 8; ++dg) {
#pragma unroll
    for (int e = 0; e < 8; ++e) {
      int qq = qWaveBase + e + 8 * hihalf;
      Og[(((size_t)b * S_ + qq) * H_ + h) * D_ + dg * 16 + l15] = acc[dg][e] * lsel[e];
    }
  }
}

extern "C" void kernel_launch(void* const* d_in, const int* in_sizes, int n_in,
                              void* d_out, int out_size, void* d_ws, size_t ws_size,
                              hipStream_t stream) {
  const float* Q = (const float*)d_in[0];
  const float* K = (const float*)d_in[1];
  const float* V = (const float*)d_in[2];
  float* O = (float*)d_out;
  dim3 grid(S_ / 128, H_, B_);
  fa_fwd_kernel<<<grid, dim3(256), 0, stream>>>(Q, K, V, O);
}
